// MultiRelationalGraphSAGE_9904194584994
// MI455X (gfx1250) — matrix operand order repaired and checked by reading, not compiled, not run on hardware
//
#include <hip/hip_runtime.h>
#include <hip/hip_bf16.h>

#define N_NODES 20000
#define N_ETYPES 5
#define N_EDGES 64000
#define IN_C  2048
#define HID_C 1024
#define OUT_C 512

typedef __attribute__((ext_vector_type(16))) __bf16 bf16x16;
typedef __attribute__((ext_vector_type(8)))  __bf16 bf16x8;
typedef __attribute__((ext_vector_type(4)))  __bf16 bf16x4;
typedef __attribute__((ext_vector_type(8)))  float  f32x8;
typedef __attribute__((ext_vector_type(4)))  float  f32x4;

// ---------------- utility kernels ----------------

__global__ void k_zero(f32x4* __restrict__ p, long n4) {
    long i = (long)blockIdx.x * blockDim.x + threadIdx.x;
    if (i < n4) { f32x4 z = {0.f, 0.f, 0.f, 0.f}; p[i] = z; }
}

__global__ void k_deg(const int* __restrict__ dst, float* __restrict__ deg) {
    int e = blockIdx.x * blockDim.x + threadIdx.x;
    if (e < N_EDGES) atomicAdd(&deg[dst[e]], 1.0f);
}

__global__ void k_recip(float* __restrict__ deg) {
    int i = blockIdx.x * blockDim.x + threadIdx.x;
    if (i < N_NODES) deg[i] = 1.0f / fmaxf(deg[i], 1.0f);
}

// msg_sum[dst] += h[src], channel-parallel (coalesced gather + coalesced atomics)
__global__ void k_scatter(const float* __restrict__ h, const int* __restrict__ src,
                          const int* __restrict__ dst, float* __restrict__ out, int cShift) {
    long t = (long)blockIdx.x * blockDim.x + threadIdx.x;
    int e = (int)(t >> cShift);
    if (e >= N_EDGES) return;
    int c = (int)(t & ((1 << cShift) - 1));
    atomicAdd(&out[((long)dst[e] << cShift) + c], h[((long)src[e] << cShift) + c]);
}

// ---------------- fused SAGE GEMM (WMMA bf16 -> f32, double-buffered) ----------------
// C[M,N] = relu?( rowscale(A1) @ W1^T + A2 @ W2^T + bias )
// W stored [N,K] row-major (reference computes h @ W.T).

#define BM 128
#define BN 128
#define BK 32
#define LDA 40   // bf16 LDS pitch (80B: 16B-aligned, bank-conflict-spreading)

__launch_bounds__(256)
__global__ void k_gemm_sage(const float* __restrict__ A1, const float* __restrict__ rscale,
                            const float* __restrict__ A2,
                            const float* __restrict__ W1, const float* __restrict__ W2,
                            const float* __restrict__ bias,
                            float* __restrict__ Cout,
                            int M, int N, int K, int doRelu)
{
    __shared__ __bf16 sA[2][BM * LDA];
    __shared__ __bf16 sB[2][BN * LDA];

    const int tid  = threadIdx.x;
    const int lane = tid & 31;
    const int wave = tid >> 5;
    const int m0 = blockIdx.y * BM;
    const int n0 = blockIdx.x * BN;
    const int wy = wave & 3;   // 4 waves along M  -> rows wy*32..+32
    const int wx = wave >> 2;  // 2 waves along N  -> cols wx*64..+64

    f32x8 acc[2][4];
#pragma unroll
    for (int mt = 0; mt < 2; ++mt)
#pragma unroll
        for (int nt = 0; nt < 4; ++nt) acc[mt][nt] = (f32x8){0.f,0.f,0.f,0.f,0.f,0.f,0.f,0.f};

    // staging slot state (hoisted, branch-free): thread covers rows p*32 + tid/8,
    // cols (tid&7)*4..+3. OOB rows: clamp address to M-1 and multiply by scale 0.
    const int lr = tid >> 3;
    const int lc = (tid & 7) * 4;

    long aoff[4];          // clamped A row offset (elements)
    long boff[4];          // W row offset (N always multiple of 128)
    float s0[4], s1[4];    // per-slot A scale for phase0 (inv-deg) / phase1 (1 or 0)
#pragma unroll
    for (int p = 0; p < 4; ++p) {
        int row   = p * 32 + lr;
        int grow  = m0 + row;
        int valid = grow < M;
        int gc    = valid ? grow : (M - 1);
        aoff[p] = (long)gc * K + lc;
        boff[p] = (long)(n0 + row) * K + lc;
        s0[p] = valid ? rscale[gc] : 0.0f;
        s1[p] = valid ? 1.0f : 0.0f;
    }

    const int TK = K / BK;   // K-tiles per phase
    const int T  = 2 * TK;   // flat tile loop over both phases

    // fragment addressing (contiguous k = 16*(lane>>4) + s, same order for A and B)
    const int arow = lane & 15;
    const int akb  = (lane >> 4) * 16;   // A: K = akb..akb+15 contiguous
    const int bkb  = (lane >> 4) * 16;   // B: K = bkb..bkb+15 contiguous

    f32x4 ra[4], rb[4];

    // ---- prologue: load tile 0 (phase 0, k=0) and stage into buffer 0 ----
#pragma unroll
    for (int p = 0; p < 4; ++p) {
        ra[p] = *(const f32x4*)(A1 + aoff[p]);
        rb[p] = *(const f32x4*)(W1 + boff[p]);
    }
#pragma unroll
    for (int p = 0; p < 4; ++p) {
        bf16x4 a, b;
#pragma unroll
        for (int j = 0; j < 4; ++j) {
            a[j] = (__bf16)(ra[p][j] * s0[p]);
            b[j] = (__bf16)(rb[p][j]);
        }
        *(bf16x4*)(&sA[0][(p * 32 + lr) * LDA + lc]) = a;
        *(bf16x4*)(&sB[0][(p * 32 + lr) * LDA + lc]) = b;
    }

    int pbuf = 0;
    for (int t = 0; t < T; ++t) {
        __syncthreads();   // staged buffer pbuf visible to all waves

        // ---- issue global loads for tile t+1 (overlap with WMMA below) ----
        const bool have_next = (t + 1) < T;
        int nph = 0;
        if (have_next) {
            int tn = t + 1;
            nph = (tn >= TK);
            int kk = (tn - (nph ? TK : 0)) * BK;
            const float* A = nph ? A2 : A1;
            const float* W = nph ? W2 : W1;
#pragma unroll
            for (int p = 0; p < 4; ++p) {
                ra[p] = *(const f32x4*)(A + aoff[p] + kk);
                rb[p] = *(const f32x4*)(W + boff[p] + kk);
            }
        }

        // ---- fragments + 8 back-to-back WMMAs ----
        bf16x16 af[2], bfv[4];
#pragma unroll
        for (int mt = 0; mt < 2; ++mt) {
            const __bf16* base = &sA[pbuf][(wy * 32 + mt * 16 + arow) * LDA];
            bf16x8 lo = *(const bf16x8*)(base + akb);
            bf16x8 hi = *(const bf16x8*)(base + akb + 8);
            af[mt] = __builtin_shufflevector(lo, hi,
                0,1,2,3,4,5,6,7,8,9,10,11,12,13,14,15);
        }
#pragma unroll
        for (int nt = 0; nt < 4; ++nt)
            bfv[nt] = *(const bf16x16*)(&sB[pbuf][(wx * 64 + nt * 16 + arow) * LDA + bkb]);
#pragma unroll
        for (int nt = 0; nt < 4; ++nt)
#pragma unroll
            for (int mt = 0; mt < 2; ++mt)
                acc[mt][nt] = __builtin_amdgcn_wmma_f32_16x16x32_bf16(
                    false, af[mt], false, bfv[nt],
                    (short)0, acc[mt][nt], false, false);

        // ---- stage tile t+1 into the other buffer ----
        if (have_next) {
            __syncthreads();   // all waves done reading buffer pbuf^1 (from t-1)
#pragma unroll
            for (int p = 0; p < 4; ++p) {
                float s = nph ? s1[p] : s0[p];
                bf16x4 a, b;
#pragma unroll
                for (int j = 0; j < 4; ++j) {
                    a[j] = (__bf16)(ra[p][j] * s);
                    b[j] = (__bf16)(rb[p][j]);
                }
                *(bf16x4*)(&sA[pbuf ^ 1][(p * 32 + lr) * LDA + lc]) = a;
                *(bf16x4*)(&sB[pbuf ^ 1][(p * 32 + lr) * LDA + lc]) = b;
            }
            pbuf ^= 1;
        }
    }

    // ---- epilogue: C/D layout -> lane col = lane&15, rows = v + (lane>>4)*8 ----
#pragma unroll
    for (int mt = 0; mt < 2; ++mt)
#pragma unroll
        for (int nt = 0; nt < 4; ++nt) {
            int col   = n0 + wx * 64 + nt * 16 + (lane & 15);
            int rbase = m0 + wy * 32 + mt * 16 + (lane >> 4) * 8;
            float bv = bias ? bias[col] : 0.0f;
#pragma unroll
            for (int v = 0; v < 8; ++v) {
                int row = rbase + v;
                if (row < M) {
                    float val = acc[mt][nt][v] + bv;
                    if (doRelu) val = fmaxf(val, 0.0f);
                    Cout[(long)row * N + col] = val;
                }
            }
        }
}

// ---------------- softmax-weighted combine ----------------
__global__ void k_combine(const float* __restrict__ h3, const float* __restrict__ attn,
                          float* __restrict__ accum, int typeIdx) {
    long i = (long)blockIdx.x * blockDim.x + threadIdx.x;
    if (i >= (long)N_NODES * OUT_C) return;
    float a[N_ETYPES], m = -1e30f;
#pragma unroll
    for (int t = 0; t < N_ETYPES; ++t) { a[t] = attn[t]; m = fmaxf(m, a[t]); }
    float s = 0.f;
#pragma unroll
    for (int t = 0; t < N_ETYPES; ++t) s += __expf(a[t] - m);
    float w = __expf(a[typeIdx] - m) / s;
    float v = w * h3[i];
    accum[i] = (typeIdx == 0) ? v : (accum[i] + v);
}

// ---------------- per-row L2 normalize ----------------
__global__ void k_l2norm(const float* __restrict__ in, float* __restrict__ out) {
    __shared__ float red[256];
    int row = blockIdx.x;
    const float* r = in + (long)row * OUT_C;
    float s = 0.f;
    for (int c = threadIdx.x; c < OUT_C; c += blockDim.x) { float v = r[c]; s += v * v; }
    red[threadIdx.x] = s;
    __syncthreads();
    for (int off = 128; off > 0; off >>= 1) {
        if (threadIdx.x < off) red[threadIdx.x] += red[threadIdx.x + off];
        __syncthreads();
    }
    float inv = 1.0f / fmaxf(sqrtf(red[0]), 1e-12f);
    for (int c = threadIdx.x; c < OUT_C; c += blockDim.x)
        out[(long)row * OUT_C + c] = r[c] * inv;
}

// ---------------- host orchestration ----------------
extern "C" void kernel_launch(void* const* d_in, const int* in_sizes, int n_in,
                              void* d_out, int out_size, void* d_ws, size_t ws_size,
                              hipStream_t stream) {
    const float* x     = (const float*)d_in[0];
    const int*   edges = (const int*)d_in[1];
    const float* Wl1 = (const float*)d_in[2];
    const float* bl1 = (const float*)d_in[3];
    const float* Wr1 = (const float*)d_in[4];
    const float* Wl2 = (const float*)d_in[5];
    const float* bl2 = (const float*)d_in[6];
    const float* Wr2 = (const float*)d_in[7];
    const float* Wl3 = (const float*)d_in[8];
    const float* bl3 = (const float*)d_in[9];
    const float* Wr3 = (const float*)d_in[10];
    const float* attn = (const float*)d_in[11];
    float* out = (float*)d_out;

    float* ws   = (float*)d_ws;
    float* degb = ws;                                   // 20480 (padded)
    float* aggb = degb + 20480;                         // N_NODES*IN_C
    float* hAb  = aggb + (long)N_NODES * IN_C;          // N_NODES*HID_C
    float* hBb  = hAb + (long)N_NODES * HID_C;          // N_NODES*HID_C
    float* oacc = hBb + (long)N_NODES * HID_C;          // N_NODES*OUT_C

    const dim3 blk(256);
    const int  gy = (N_NODES + BM - 1) / BM;            // 157

    for (int i = 0; i < N_ETYPES; ++i) {
        const int* src = edges + (long)i * 2 * N_EDGES;
        const int* dst = src + N_EDGES;
        const float* wl1 = Wl1 + (long)i * HID_C * IN_C;
        const float* wr1 = Wr1 + (long)i * HID_C * IN_C;
        const float* wl2 = Wl2 + (long)i * HID_C * HID_C;
        const float* wr2 = Wr2 + (long)i * HID_C * HID_C;
        const float* wl3 = Wl3 + (long)i * OUT_C * HID_C;
        const float* wr3 = Wr3 + (long)i * OUT_C * HID_C;

        // degree -> 1/max(deg,1)
        k_zero<<<20, blk, 0, stream>>>((f32x4*)degb, 20480 / 4);
        k_deg<<<(N_EDGES + 255) / 256, blk, 0, stream>>>(dst, degb);
        k_recip<<<(N_NODES + 255) / 256, blk, 0, stream>>>(degb);

        // layer 1: C_in = 2048
        k_zero<<<(int)(((long)N_NODES * IN_C / 4 + 255) / 256), blk, 0, stream>>>(
            (f32x4*)aggb, (long)N_NODES * IN_C / 4);
        k_scatter<<<(int)(((long)N_EDGES * IN_C + 255) / 256), blk, 0, stream>>>(
            x, src, dst, aggb, 11);
        k_gemm_sage<<<dim3(HID_C / BN, gy), blk, 0, stream>>>(
            aggb, degb, x, wl1, wr1, bl1 + (long)i * HID_C, hAb,
            N_NODES, HID_C, IN_C, 1);

        // layer 2: C = 1024
        k_zero<<<(int)(((long)N_NODES * HID_C / 4 + 255) / 256), blk, 0, stream>>>(
            (f32x4*)aggb, (long)N_NODES * HID_C / 4);
        k_scatter<<<(int)(((long)N_EDGES * HID_C + 255) / 256), blk, 0, stream>>>(
            hAb, src, dst, aggb, 10);
        k_gemm_sage<<<dim3(HID_C / BN, gy), blk, 0, stream>>>(
            aggb, degb, hAb, wl2, wr2, bl2 + (long)i * HID_C, hBb,
            N_NODES, HID_C, HID_C, 1);

        // layer 3: C = 1024 -> 512, no relu
        k_zero<<<(int)(((long)N_NODES * HID_C / 4 + 255) / 256), blk, 0, stream>>>(
            (f32x4*)aggb, (long)N_NODES * HID_C / 4);
        k_scatter<<<(int)(((long)N_EDGES * HID_C + 255) / 256), blk, 0, stream>>>(
            hBb, src, dst, aggb, 10);
        k_gemm_sage<<<dim3(OUT_C / BN, gy), blk, 0, stream>>>(
            aggb, degb, hBb, wl3, wr3, bl3 + (long)i * OUT_C, hAb,
            N_NODES, OUT_C, HID_C, 0);

        // softmax-weighted accumulate
        k_combine<<<(int)(((long)N_NODES * OUT_C + 255) / 256), blk, 0, stream>>>(
            hAb, attn, oacc, i);
    }

    k_l2norm<<<N_NODES, blk, 0, stream>>>(oacc, out);
}
